// Mamba2_40303973106194
// MI455X (gfx1250) — matrix operand order repaired and checked by reading, not compiled, not run on hardware
//
#include <hip/hip_runtime.h>
#include <hip/hip_bf16.h>
#include <math.h>

// ---------------- constants ----------------
#define D_MODEL   2048
#define D_INNER   4096
#define D_SSM     4096
#define NHEADS    64
#define HEADDIM   64
#define D_STATE   128
#define D_CONV    4
#define CHUNK     256
#define NCHUNK    8
#define B_SZ      2
#define SEQLEN    2048
#define D_IN_PROJ 8512
#define CONV_DIM  4352
#define NTOK      (B_SZ*SEQLEN)   // 4096

typedef __bf16 v16bf __attribute__((ext_vector_type(16)));
typedef __bf16 v8bf  __attribute__((ext_vector_type(8)));
typedef float  v8f   __attribute__((ext_vector_type(8)));

__device__ __forceinline__ v16bf frag_cat(v8bf lo, v8bf hi) {
    return __builtin_shufflevector(lo, hi, 0,1,2,3,4,5,6,7,8,9,10,11,12,13,14,15);
}

// =====================================================================
// WMMA bf16 GEMM, 128x128 block tile, double-buffered LDS, one barrier
// per K-step.  C[M,N] = A[M,K] * B[K,N]  (f32 memory, bf16 math, f32 acc)
// BT=true: B given row-major [N,K]  (computes A*B^T).
// Requires M%128==0, K%32==0, N%4==0 (N tail tile handled by clamp+guard).
// 256 threads = 8 waves; wave (wm=w&3, wn=w>>2) owns 32x64 -> 2x4 WMMA.
// =====================================================================
template <bool BT>
__global__ __launch_bounds__(256) void wmma_gemm128(
    const float* __restrict__ A, const float* __restrict__ B,
    float* __restrict__ C, int M, int N, int K,
    int lda, int ldb, int ldc,
    long long strideA, long long strideB, long long strideC)
{
    A += (long long)blockIdx.z * strideA;
    B += (long long)blockIdx.z * strideB;
    C += (long long)blockIdx.z * strideC;

    const int n0   = blockIdx.x * 128;
    const int m0   = blockIdx.y * 128;
    const int tid  = threadIdx.x;
    const int lane = tid & 31;
    const int w    = tid >> 5;
    const int wm   = w & 3;            // 4 m-positions * 32 rows
    const int wn   = w >> 2;           // 2 n-positions * 64 cols
    const int lr   = lane & 15;
    const int half = lane >> 4;

    // double-buffered tiles; rows padded to 40 (80B) keeps 16B alignment
    __shared__ __attribute__((aligned(16))) __bf16 As[2][128][40];  // [m][k]
    __shared__ __attribute__((aligned(16))) __bf16 Bs[2][128][40];  // [n][k]

    v8f acc[2][4];
    for (int i = 0; i < 2; ++i)
        for (int j = 0; j < 4; ++j)
            for (int e = 0; e < 8; ++e) acc[i][j][e] = 0.f;

    float4 areg[4], breg[4];
    auto load_regs = [&](int kb) {
#pragma unroll
        for (int i = 0; i < 4; ++i) {               // A tile 128x32
            int idx = tid + i * 256;
            int r   = idx >> 3;
            int c4  = (idx & 7) << 2;
            areg[i] = *(const float4*)(A + (long long)(m0 + r) * lda + kb + c4);
        }
#pragma unroll
        for (int i = 0; i < 4; ++i) {
            int idx = tid + i * 256;
            if (BT) {                               // B tile [n][k] 128x32
                int r  = idx >> 3;
                int c4 = (idx & 7) << 2;
                int rr = n0 + r; if (rr > N - 1) rr = N - 1;
                breg[i] = *(const float4*)(B + (long long)rr * ldb + kb + c4);
            } else {                                // B tile [k][n] 32x128
                int r  = idx >> 5;
                int c4 = (idx & 31) << 2;
                int cc = n0 + c4; if (cc > N - 4) cc = N - 4;   // clamp tail
                breg[i] = *(const float4*)(B + (long long)(kb + r) * ldb + cc);
            }
        }
    };
    auto store_lds = [&](int buf) {
#pragma unroll
        for (int i = 0; i < 4; ++i) {
            int idx = tid + i * 256;
            int r   = idx >> 3;
            int c4  = (idx & 7) << 2;
            As[buf][r][c4 + 0] = (__bf16)areg[i].x;
            As[buf][r][c4 + 1] = (__bf16)areg[i].y;
            As[buf][r][c4 + 2] = (__bf16)areg[i].z;
            As[buf][r][c4 + 3] = (__bf16)areg[i].w;
        }
#pragma unroll
        for (int i = 0; i < 4; ++i) {
            int idx = tid + i * 256;
            if (BT) {
                int r  = idx >> 3;
                int c4 = (idx & 7) << 2;
                Bs[buf][r][c4 + 0] = (__bf16)breg[i].x;
                Bs[buf][r][c4 + 1] = (__bf16)breg[i].y;
                Bs[buf][r][c4 + 2] = (__bf16)breg[i].z;
                Bs[buf][r][c4 + 3] = (__bf16)breg[i].w;
            } else {
                int r  = idx >> 5;
                int c4 = (idx & 31) << 2;
                Bs[buf][c4 + 0][r] = (__bf16)breg[i].x;   // transpose to [n][k]
                Bs[buf][c4 + 1][r] = (__bf16)breg[i].y;
                Bs[buf][c4 + 2][r] = (__bf16)breg[i].z;
                Bs[buf][c4 + 3][r] = (__bf16)breg[i].w;
            }
        }
    };

    // prologue: stage K-step 0
    load_regs(0);
    store_lds(0);
    __syncthreads();

    int cur = 0;
    for (int k0 = 0; k0 < K; k0 += 32) {
        const bool more = (k0 + 32 < K);
        if (more) load_regs(k0 + 32);   // overlap next-tile loads with WMMA

        v16bf afr[2], bfr[4];
#pragma unroll
        for (int mi = 0; mi < 2; ++mi) {
            int row = wm * 32 + mi * 16 + lr;
            afr[mi] = frag_cat(*(const v8bf*)&As[cur][row][half * 8],
                               *(const v8bf*)&As[cur][row][16 + half * 8]);
        }
#pragma unroll
        for (int nj = 0; nj < 4; ++nj) {
            int col = wn * 64 + nj * 16 + lr;
            bfr[nj] = frag_cat(*(const v8bf*)&Bs[cur][col][half * 8],
                               *(const v8bf*)&Bs[cur][col][16 + half * 8]);
        }
#pragma unroll
        for (int mi = 0; mi < 2; ++mi)
#pragma unroll
            for (int nj = 0; nj < 4; ++nj)
                acc[mi][nj] = __builtin_amdgcn_wmma_f32_16x16x32_bf16(
                    false, afr[mi], false, bfr[nj], (short)0, acc[mi][nj], false, false);

        if (more) {
            store_lds(cur ^ 1);
            __syncthreads();            // single barrier per K-step
            cur ^= 1;
        }
    }

#pragma unroll
    for (int mi = 0; mi < 2; ++mi)
#pragma unroll
        for (int nj = 0; nj < 4; ++nj) {
            int gm0 = m0 + wm * 32 + mi * 16 + half * 8;
            int gn  = n0 + wn * 64 + nj * 16 + lr;
            if (gn < N) {
#pragma unroll
                for (int r = 0; r < 8; ++r)
                    C[(long long)(gm0 + r) * ldc + gn] = acc[mi][nj][r];
            }
        }
}

// =====================================================================
// depthwise causal conv (width 4) + bias + SiLU
// =====================================================================
__global__ __launch_bounds__(256) void conv_silu_k(
    const float* __restrict__ zx, const float* __restrict__ cw,
    const float* __restrict__ cb, float* __restrict__ xbc)
{
    long long idx = (long long)blockIdx.x * 256 + threadIdx.x;
    int c = (int)(idx % CONV_DIM);
    long long tok = idx / CONV_DIM;
    int l = (int)(tok % SEQLEN);
    float acc = cb[c];
#pragma unroll
    for (int j = 0; j < D_CONV; ++j) {
        int tl = l - (D_CONV - 1) + j;
        if (tl >= 0)
            acc += zx[(tok - l + tl) * D_IN_PROJ + D_SSM + c] * cw[c * D_CONV + j];
    }
    acc = acc / (1.f + __expf(-acc));
    xbc[idx] = acc;
}

__global__ __launch_bounds__(256) void prep_dt_k(
    const float* __restrict__ zx, const float* __restrict__ dt_bias,
    const float* __restrict__ A_log, float* __restrict__ dt, float* __restrict__ dA)
{
    int idx = blockIdx.x * 256 + threadIdx.x;
    int h = idx & 63;
    long long tok = idx >> 6;
    float x = zx[tok * D_IN_PROJ + (D_IN_PROJ - NHEADS) + h] + dt_bias[h];
    float sp = (x > 20.f) ? x : log1pf(__expf(x));
    dt[idx] = sp;
    dA[idx] = -__expf(A_log[h]) * sp;
}

__global__ __launch_bounds__(256) void xscale_k(
    const float* __restrict__ xbc, const float* __restrict__ dt, float* __restrict__ Xd)
{
    long long idx = (long long)blockIdx.x * 256 + threadIdx.x;
    long long tok = idx >> 12;
    int hp = (int)(idx & 4095);
    Xd[idx] = xbc[tok * CONV_DIM + hp] * dt[tok * NHEADS + (hp >> 6)];
}

__global__ void cumsum_k(const float* __restrict__ dA, float* __restrict__ Acum)
{
    int z = blockIdx.x;
    int h = threadIdx.x;
    int b = z >> 3, c = z & 7;
    const float* src = dA + ((size_t)b * SEQLEN + c * CHUNK) * NHEADS + h;
    float* dst = Acum + ((size_t)z * NHEADS + h) * CHUNK;
    float run = 0.f;
    for (int t = 0; t < CHUNK; ++t) { run += src[(size_t)t * NHEADS]; dst[t] = run; }
}

// =====================================================================
// per-chunk states via WMMA:  S[p,n] = sum_t (Xd[t,p]*dec[t]) * B[t,n]
// one block per (b,c,h): M=64(p), N=128(n), K=256(t). 8 waves, 32x32/wave.
// =====================================================================
__global__ __launch_bounds__(256) void chunk_states_wmma(
    const float* __restrict__ xbc, const float* __restrict__ Xd,
    const float* __restrict__ Acum, float* __restrict__ cst)
{
    int bid = blockIdx.x;                 // B*8*64 = 1024
    int h = bid & 63, c = (bid >> 6) & 7, b = bid >> 9;
    int tid  = threadIdx.x;
    int lane = tid & 31;
    int w    = tid >> 5;
    int wm   = w >> 2;                    // 0..1 over p
    int wn   = w & 3;                     // 0..3 over n
    int lr   = lane & 15, half = lane >> 4;

    __shared__ __attribute__((aligned(16))) __bf16 As_[64][40];   // [p][t]
    __shared__ __attribute__((aligned(16))) __bf16 Bs_[128][40];  // [n][t]
    __shared__ float dec[CHUNK];

    size_t z = (size_t)(b * 8 + c);
    const float* Ac = Acum + (z * NHEADS + h) * CHUNK;
    float alast = Ac[CHUNK - 1];
    dec[tid] = __expf(alast - Ac[tid]);

    size_t tokbase = (size_t)b * SEQLEN + c * CHUNK;
    const float* Xb = Xd + tokbase * D_SSM + h * HEADDIM;
    const float* Bb = xbc + tokbase * CONV_DIM + D_SSM;

    v8f acc[2][2];
    for (int i = 0; i < 2; ++i)
        for (int j = 0; j < 2; ++j)
            for (int e = 0; e < 8; ++e) acc[i][j][e] = 0.f;

    for (int t0 = 0; t0 < CHUNK; t0 += 32) {
        __syncthreads();
#pragma unroll
        for (int i = 0; i < 8; ++i) {
            int idx = tid + i * 256;
            int tt = idx >> 6, p = idx & 63;
            As_[p][tt] = (__bf16)(Xb[(size_t)(t0 + tt) * D_SSM + p] * dec[t0 + tt]);
        }
#pragma unroll
        for (int i = 0; i < 16; ++i) {
            int idx = tid + i * 256;
            int tt = idx >> 7, n = idx & 127;
            Bs_[n][tt] = (__bf16)Bb[(size_t)(t0 + tt) * CONV_DIM + n];
        }
        __syncthreads();
        v16bf afr[2], bfr[2];
#pragma unroll
        for (int mi = 0; mi < 2; ++mi) {
            int row = wm * 32 + mi * 16 + lr;
            afr[mi] = frag_cat(*(const v8bf*)&As_[row][half * 8],
                               *(const v8bf*)&As_[row][16 + half * 8]);
        }
#pragma unroll
        for (int nj = 0; nj < 2; ++nj) {
            int col = wn * 32 + nj * 16 + lr;
            bfr[nj] = frag_cat(*(const v8bf*)&Bs_[col][half * 8],
                               *(const v8bf*)&Bs_[col][16 + half * 8]);
        }
#pragma unroll
        for (int mi = 0; mi < 2; ++mi)
#pragma unroll
            for (int nj = 0; nj < 2; ++nj)
                acc[mi][nj] = __builtin_amdgcn_wmma_f32_16x16x32_bf16(
                    false, afr[mi], false, bfr[nj], (short)0, acc[mi][nj], false, false);
    }

    float* o = cst + (z * NHEADS + h) * (HEADDIM * D_STATE);
#pragma unroll
    for (int mi = 0; mi < 2; ++mi)
#pragma unroll
        for (int nj = 0; nj < 2; ++nj) {
            int gp0 = wm * 32 + mi * 16 + half * 8;
            int gn  = wn * 32 + nj * 16 + lr;
#pragma unroll
            for (int r = 0; r < 8; ++r)
                o[(size_t)(gp0 + r) * D_STATE + gn] = acc[mi][nj][r];
        }
}

// sequential inter-chunk scan (tiny)
__global__ __launch_bounds__(256) void chunk_scan_k(
    const float* __restrict__ Acum, const float* __restrict__ cst,
    float* __restrict__ stIn)
{
    int h = blockIdx.x & 63, b = blockIdx.x >> 6;
    int tid = threadIdx.x;
    int n = tid & 127, pb = tid >> 7;
    float S[32];
#pragma unroll
    for (int i = 0; i < 32; ++i) S[i] = 0.f;
    for (int c = 0; c < NCHUNK; ++c) {
        size_t zb = ((size_t)(b * 8 + c) * NHEADS + h);
        float g = __expf(Acum[zb * CHUNK + (CHUNK - 1)]);
        size_t base = zb * (HEADDIM * D_STATE) + n;
#pragma unroll
        for (int i = 0; i < 32; ++i) {
            size_t o = base + (size_t)(pb + 2 * i) * D_STATE;
            stIn[o] = S[i];
            S[i] = S[i] * g + cst[o];
        }
    }
}

// =====================================================================
// Fused Y = Y_diag + Y_off + D*x via WMMA.  One block per (b,c,h).
// K = 256(s, masked decayed CB) + 128(n, e^{A}C x S^T) = 12 slabs of 32.
// =====================================================================
__global__ __launch_bounds__(256) void ssd_y_wmma(
    const float* __restrict__ CBL, const float* __restrict__ Acum,
    const float* __restrict__ Xd, const float* __restrict__ xbc,
    const float* __restrict__ stIn, const float* __restrict__ Dp,
    float* __restrict__ Yraw)
{
    int bid = blockIdx.x;                 // B*8*64 = 1024
    int h = bid & 63, c = (bid >> 6) & 7, b = bid >> 9;
    int tid  = threadIdx.x;
    int lane = tid & 31;
    int w    = tid >> 5;                  // wave 0..7 -> t rows
    int lr   = lane & 15, half = lane >> 4;

    __shared__ __attribute__((aligned(16))) __bf16 Aw[CHUNK][40];  // [t][k-slab]
    __shared__ __attribute__((aligned(16))) __bf16 Bx[64][40];     // [p][k-slab]
    __shared__ float At[CHUNK];
    __shared__ float eAt[CHUNK];

    size_t z = (size_t)(b * 8 + c);
    const float* Ac = Acum + (z * NHEADS + h) * CHUNK;
    float at = Ac[tid];
    At[tid]  = at;
    eAt[tid] = __expf(at);

    size_t tokbase = (size_t)b * SEQLEN + c * CHUNK;
    const float* cbrow = CBL + (z * CHUNK + tid) * CHUNK;
    const float* crow  = xbc + (tokbase + tid) * CONV_DIM + D_SSM + D_STATE;
    const float* Xb    = Xd + tokbase * D_SSM + h * HEADDIM;
    const float* st    = stIn + (z * NHEADS + h) * (HEADDIM * D_STATE);

    v8f acc[2][4];
    for (int i = 0; i < 2; ++i)
        for (int j = 0; j < 4; ++j)
            for (int e = 0; e < 8; ++e) acc[i][j][e] = 0.f;

    for (int kk = 0; kk < 12; ++kk) {
        __syncthreads();
        if (kk < 8) {
            int s0 = kk * 32;
#pragma unroll
            for (int ss = 0; ss < 32; ++ss) {
                int s = s0 + ss;
                float v = (s <= tid) ? cbrow[s] * __expf(at - At[s]) : 0.f;
                Aw[tid][ss] = (__bf16)v;
            }
#pragma unroll
            for (int i = 0; i < 8; ++i) {
                int idx = tid + i * 256;
                int ss = idx >> 6, p = idx & 63;
                Bx[p][ss] = (__bf16)Xb[(size_t)(s0 + ss) * D_SSM + p];
            }
        } else {
            int n0 = (kk - 8) * 32;
            float eat = eAt[tid];
#pragma unroll
            for (int nn = 0; nn < 32; ++nn)
                Aw[tid][nn] = (__bf16)(eat * crow[n0 + nn]);
#pragma unroll
            for (int i = 0; i < 8; ++i) {
                int idx = tid + i * 256;
                int nn = idx & 31, p = idx >> 5;
                Bx[p][nn] = (__bf16)st[(size_t)p * D_STATE + n0 + nn];
            }
        }
        __syncthreads();
        v16bf afr[2], bfr[4];
#pragma unroll
        for (int mi = 0; mi < 2; ++mi) {
            int row = w * 32 + mi * 16 + lr;
            afr[mi] = frag_cat(*(const v8bf*)&Aw[row][half * 8],
                               *(const v8bf*)&Aw[row][16 + half * 8]);
        }
#pragma unroll
        for (int nj = 0; nj < 4; ++nj) {
            int col = nj * 16 + lr;
            bfr[nj] = frag_cat(*(const v8bf*)&Bx[col][half * 8],
                               *(const v8bf*)&Bx[col][16 + half * 8]);
        }
#pragma unroll
        for (int mi = 0; mi < 2; ++mi)
#pragma unroll
            for (int nj = 0; nj < 4; ++nj)
                acc[mi][nj] = __builtin_amdgcn_wmma_f32_16x16x32_bf16(
                    false, afr[mi], false, bfr[nj], (short)0, acc[mi][nj], false, false);
    }

    float dpv = Dp[h];
#pragma unroll
    for (int mi = 0; mi < 2; ++mi)
#pragma unroll
        for (int nj = 0; nj < 4; ++nj) {
            int gt0 = w * 32 + mi * 16 + half * 8;
            int gp  = nj * 16 + lr;
#pragma unroll
            for (int r = 0; r < 8; ++r) {
                size_t row = tokbase + gt0 + r;
                Yraw[row * D_SSM + h * HEADDIM + gp] =
                    acc[mi][nj][r] + dpv * xbc[row * CONV_DIM + h * HEADDIM + gp];
            }
        }
}

// Y *= silu(z); RMSNorm with norm_w
__global__ __launch_bounds__(256) void gate_rmsnorm_k(
    const float* __restrict__ Yraw, const float* __restrict__ zx,
    const float* __restrict__ norm_w, float* __restrict__ Yn)
{
    int tok = blockIdx.x, tid = threadIdx.x;
    __shared__ float red[256];
    const float* yr = Yraw + (size_t)tok * D_SSM;
    const float* zr = zx + (size_t)tok * D_IN_PROJ;
    float yv[16];
    float local = 0.f;
#pragma unroll
    for (int i = 0; i < 16; ++i) {
        int col = tid + i * 256;
        float zv = zr[col];
        float y = yr[col] * (zv / (1.f + __expf(-zv)));
        yv[i] = y;
        local += y * y;
    }
    red[tid] = local;
    __syncthreads();
    for (int s = 128; s > 0; s >>= 1) {
        if (tid < s) red[tid] += red[tid + s];
        __syncthreads();
    }
    float scale = rsqrtf(red[0] * (1.f / (float)D_SSM) + 1e-5f);
    float* o = Yn + (size_t)tok * D_SSM;
#pragma unroll
    for (int i = 0; i < 16; ++i) {
        int col = tid + i * 256;
        o[col] = yv[i] * scale * norm_w[col];
    }
}

// =====================================================================
extern "C" void kernel_launch(void* const* d_in, const int* in_sizes, int n_in,
                              void* d_out, int out_size, void* d_ws, size_t ws_size,
                              hipStream_t stream)
{
    (void)in_sizes; (void)n_in; (void)out_size; (void)ws_size;
    const float* u       = (const float*)d_in[0];
    const float* W_in    = (const float*)d_in[1];
    const float* conv_w  = (const float*)d_in[2];
    const float* conv_b  = (const float*)d_in[3];
    const float* dt_bias = (const float*)d_in[4];
    const float* A_log   = (const float*)d_in[5];
    const float* Dp      = (const float*)d_in[6];
    const float* norm_w  = (const float*)d_in[7];
    const float* W_out   = (const float*)d_in[8];
    float* out = (float*)d_out;
    float* ws  = (float*)d_ws;

    size_t o = 0;
    float* zx   = ws + o; o += (size_t)NTOK * D_IN_PROJ;
    float* xbc  = ws + o; o += (size_t)NTOK * CONV_DIM;
    float* Xd   = ws + o; o += (size_t)NTOK * D_SSM;
    float* dt   = ws + o; o += (size_t)NTOK * NHEADS;
    float* dA   = ws + o; o += (size_t)NTOK * NHEADS;
    float* Acum = ws + o; o += (size_t)B_SZ * NCHUNK * NHEADS * CHUNK;
    float* CBL  = ws + o; o += (size_t)B_SZ * NCHUNK * CHUNK * CHUNK;
    float* cst  = ws + o; o += (size_t)B_SZ * NCHUNK * NHEADS * HEADDIM * D_STATE;
    float* stIn = ws + o; o += (size_t)B_SZ * NCHUNK * NHEADS * HEADDIM * D_STATE;
    float* Yraw = ws + o; o += (size_t)NTOK * D_SSM;
    float* Ynorm = xbc;   // xbc dead after ssd_y_wmma; reuse with tight stride

    // 1) zxbcdt = u @ W_in          (WMMA, 128x128 double-buffered; N tail guarded)
    wmma_gemm128<false><<<dim3((D_IN_PROJ + 127) / 128, NTOK / 128, 1), 256, 0, stream>>>(
        u, W_in, zx, NTOK, D_IN_PROJ, D_MODEL, D_MODEL, D_IN_PROJ, D_IN_PROJ, 0, 0, 0);

    // 2) depthwise conv + SiLU
    conv_silu_k<<<(NTOK * (size_t)CONV_DIM) / 256, 256, 0, stream>>>(zx, conv_w, conv_b, xbc);

    // 3) dt / dA, X*dt, per-chunk cumsum
    prep_dt_k<<<(NTOK * NHEADS) / 256, 256, 0, stream>>>(zx, dt_bias, A_log, dt, dA);
    xscale_k<<<(NTOK * (size_t)D_SSM) / 256, 256, 0, stream>>>(xbc, dt, Xd);
    cumsum_k<<<B_SZ * NCHUNK, NHEADS, 0, stream>>>(dA, Acum);

    // 4) CB[t,s] = C @ B^T per (b,chunk)   (batched WMMA, BT variant)
    wmma_gemm128<true><<<dim3(CHUNK / 128, CHUNK / 128, B_SZ * NCHUNK), 256, 0, stream>>>(
        xbc + D_SSM + D_STATE, xbc + D_SSM, CBL,
        CHUNK, CHUNK, D_STATE, CONV_DIM, CONV_DIM, CHUNK,
        (long long)CHUNK * CONV_DIM, (long long)CHUNK * CONV_DIM,
        (long long)CHUNK * CHUNK);

    // 5) per-chunk states (WMMA), inter-chunk scan
    chunk_states_wmma<<<B_SZ * NCHUNK * NHEADS, 256, 0, stream>>>(xbc, Xd, Acum, cst);
    chunk_scan_k<<<B_SZ * NHEADS, 256, 0, stream>>>(Acum, cst, stIn);

    // 6) Y = Y_diag + Y_off + D*x   (fused WMMA)
    ssd_y_wmma<<<B_SZ * NCHUNK * NHEADS, 256, 0, stream>>>(
        CBL, Acum, Xd, xbc, stIn, Dp, Yraw);

    // 7) gate + RMSNorm
    gate_rmsnorm_k<<<NTOK, 256, 0, stream>>>(Yraw, zx, norm_w, Ynorm);

    // 8) out = Ynorm @ W_out        (WMMA)
    wmma_gemm128<false><<<dim3(D_MODEL / 128, NTOK / 128, 1), 256, 0, stream>>>(
        Ynorm, W_out, out, NTOK, D_MODEL, D_INNER, D_INNER, D_MODEL, D_MODEL, 0, 0, 0);
}
